// GatedLinearRNN_25022479466754
// MI455X (gfx1250) — matrix operand order repaired and checked by reading, not compiled, not run on hardware
//
#include <hip/hip_runtime.h>
#include <hip/hip_bf16.h>

#define B_DIM 4
#define T_DIM 4096
#define D_DIM 2048
#define S_DIM 2048
#define M_DIM (B_DIM * T_DIM)      // 16384
#define BK 32
#define NCHUNK 8
#define CLEN (T_DIM / NCHUNK)      // 512

typedef __bf16 bf16;
typedef __attribute__((ext_vector_type(16))) __bf16 v16bf;
typedef __attribute__((ext_vector_type(8)))  __bf16 v8bf;
typedef __attribute__((ext_vector_type(4)))  __bf16 v4bf;
typedef __attribute__((ext_vector_type(8)))  float  v8f;
typedef __attribute__((ext_vector_type(4)))  int    v4i;

union Frag16 { v16bf v; v8bf h[2]; };

// ---- CDNA5 async global->LDS copy (ASYNCcnt), with fallback to reg staging
#if __has_builtin(__builtin_amdgcn_global_load_async_to_lds_b128)
#define HAVE_ASYNC_LDS 1
typedef __attribute__((address_space(1))) v4i gv4i;   // global 16B vector
typedef __attribute__((address_space(3))) v4i lv4i;   // LDS 16B vector
__device__ __forceinline__ void async_cp_b128(void* lds_dst, const void* gsrc) {
    __builtin_amdgcn_global_load_async_to_lds_b128((gv4i*)gsrc, (lv4i*)lds_dst, 0, 0);
}
__device__ __forceinline__ void async_wait0() {
#if __has_builtin(__builtin_amdgcn_s_wait_asynccnt)
    __builtin_amdgcn_s_wait_asynccnt(0);
#else
    asm volatile("s_wait_asynccnt 0x0" ::: "memory");
#endif
}
#else
#define HAVE_ASYNC_LDS 0
#endif

// ---------------------------------------------------------------- weight cvt
__global__ void cvt_f32_bf16(const float* __restrict__ in, bf16* __restrict__ out) {
    int i = (blockIdx.x * 256 + threadIdx.x) * 4;
    float4 f = *(const float4*)(in + i);
    v4bf o = { (bf16)f.x, (bf16)f.y, (bf16)f.z, (bf16)f.w };
    *(v4bf*)(out + i) = o;
}

// ---------------------------------------------------------------- rmsnorm
__global__ void rmsnorm_kernel(const float* __restrict__ x, const float* __restrict__ w,
                               bf16* __restrict__ xn) {
    __shared__ float red[8];
    const int row = blockIdx.x;
    const float* xr = x + (size_t)row * D_DIM;
    float v[8];
    float ss = 0.f;
#pragma unroll
    for (int i = 0; i < 8; ++i) {
        v[i] = xr[threadIdx.x + i * 256];
        ss += v[i] * v[i];
    }
#pragma unroll
    for (int o = 16; o > 0; o >>= 1) ss += __shfl_xor(ss, o, 32);
    if ((threadIdx.x & 31) == 0) red[threadIdx.x >> 5] = ss;
    __syncthreads();
    float tot = 0.f;
#pragma unroll
    for (int i = 0; i < 8; ++i) tot += red[i];
    const float scale = rsqrtf(tot * (1.0f / D_DIM) + 1e-5f);
    bf16* xo = xn + (size_t)row * D_DIM;
#pragma unroll
    for (int i = 0; i < 8; ++i) {
        int c = threadIdx.x + i * 256;
        xo[c] = (bf16)(v[i] * scale * w[c]);
    }
}

// ---------------------------------------------------------------- fused WMMA GEMM
// MODE 0: out = silu(acc0) * acc1                (bf16 out)  [og]
// MODE 1: out = sigmoid(acc0 + bias[n])          (bf16 out)  [f]
// MODE 2: out = acc0 * sigmoid(acc1)             (bf16 out)  [z]
// MODE 3: out = acc0 + bias[n] + resid[m,n]      (f32 out)   [y]
// A: [M,K] bf16 row-major. B0/B1: [N,K] bf16 row-major (weight [S,D] layout).
template <int MODE, int NPROJ>
__global__ __launch_bounds__(256, 1)
void gemm_wmma(const bf16* __restrict__ A,
               const bf16* __restrict__ B0,
               const bf16* __restrict__ B1,
               const float* __restrict__ bias,
               const float* __restrict__ resid,
               void* __restrict__ Out,
               int Ncols, int K) {
    __shared__ __align__(16) bf16 As[2][128][BK + 8];
    __shared__ __align__(16) bf16 Bs[2][NPROJ][64][BK + 8];

    const int tid  = threadIdx.x;
    const int m0   = blockIdx.y * 128;
    const int n0   = blockIdx.x * 64;
    const int lane = tid & 31;
    const int wid  = tid >> 5;
    const int wm   = wid >> 1;          // 0..3  (M sub-tile of 32)
    const int wn   = wid & 1;           // 0..1  (N sub-tile of 32)
    const int lr   = lane & 15;
    const int hk   = lane >> 4;         // half-wave id
    const int lka  = hk * 16;           // A frag K chunk base
    const int lkb  = hk * 16;           // B frag K chunk base

    // per-thread copy coordinates: 16B chunks
    const int ar = tid >> 2;            // A rows [0,64) (+64 for second chunk)
    const int ac = (tid & 3) * 8;       // K-column offset in elements
    const bf16* Arow0 = A + (size_t)(m0 + ar) * K + ac;
    const bf16* Arow1 = A + (size_t)(m0 + ar + 64) * K + ac;
    const bf16* Brow0 = B0 + (size_t)(n0 + ar) * K + ac;
    const bf16* Brow1 = (NPROJ == 2) ? (B1 + (size_t)(n0 + ar) * K + ac) : nullptr;

    const int KT = K / BK;

#if HAVE_ASYNC_LDS
    // ---- CDNA5 async DMA tile feed: no staging VGPRs at all
    {
        async_cp_b128(&As[0][ar][ac],      Arow0);
        async_cp_b128(&As[0][ar + 64][ac], Arow1);
        async_cp_b128(&Bs[0][0][ar][ac],   Brow0);
        if constexpr (NPROJ == 2) async_cp_b128(&Bs[0][1][ar][ac], Brow1);
        async_wait0();
        __syncthreads();
    }
#else
    uint4 pa0, pa1, pb0, pb1;
    pa0 = *(const uint4*)(Arow0);
    pa1 = *(const uint4*)(Arow1);
    pb0 = *(const uint4*)(Brow0);
    if constexpr (NPROJ == 2) pb1 = *(const uint4*)(Brow1);
    *(uint4*)&As[0][ar][ac]      = pa0;
    *(uint4*)&As[0][ar + 64][ac] = pa1;
    *(uint4*)&Bs[0][0][ar][ac]   = pb0;
    if constexpr (NPROJ == 2) *(uint4*)&Bs[0][1][ar][ac] = pb1;
    __syncthreads();
#endif

    v8f zero8 = {0.f, 0.f, 0.f, 0.f, 0.f, 0.f, 0.f, 0.f};
    v8f acc[NPROJ][2][2];
#pragma unroll
    for (int p = 0; p < NPROJ; ++p)
#pragma unroll
        for (int ms = 0; ms < 2; ++ms)
#pragma unroll
            for (int ns = 0; ns < 2; ++ns) acc[p][ms][ns] = zero8;

    for (int kt = 0; kt < KT; ++kt) {
        const int buf = kt & 1;
        const int k1 = (kt + 1) * BK;

        if (kt + 1 < KT) {
#if HAVE_ASYNC_LDS
            async_cp_b128(&As[buf ^ 1][ar][ac],      Arow0 + k1);
            async_cp_b128(&As[buf ^ 1][ar + 64][ac], Arow1 + k1);
            async_cp_b128(&Bs[buf ^ 1][0][ar][ac],   Brow0 + k1);
            if constexpr (NPROJ == 2)
                async_cp_b128(&Bs[buf ^ 1][1][ar][ac], Brow1 + k1);
#else
            pa0 = *(const uint4*)(Arow0 + k1);
            pa1 = *(const uint4*)(Arow1 + k1);
            pb0 = *(const uint4*)(Brow0 + k1);
            if constexpr (NPROJ == 2) pb1 = *(const uint4*)(Brow1 + k1);
#endif
        }

        v16bf af[2];
#pragma unroll
        for (int ms = 0; ms < 2; ++ms) {
            const bf16* p = &As[buf][wm * 32 + ms * 16 + lr][lka];
            Frag16 u;
            u.h[0] = *(const v8bf*)p;
            u.h[1] = *(const v8bf*)(p + 8);
            af[ms] = u.v;
        }
        v16bf bfrg[NPROJ][2];
#pragma unroll
        for (int pp = 0; pp < NPROJ; ++pp)
#pragma unroll
            for (int ns = 0; ns < 2; ++ns) {
                const bf16* p = &Bs[buf][pp][wn * 32 + ns * 16 + lr][lkb];
                Frag16 u;
                u.h[0] = *(const v8bf*)p;
                u.h[1] = *(const v8bf*)(p + 8);
                bfrg[pp][ns] = u.v;
            }
#pragma unroll
        for (int pp = 0; pp < NPROJ; ++pp)
#pragma unroll
            for (int ms = 0; ms < 2; ++ms)
#pragma unroll
                for (int ns = 0; ns < 2; ++ns)
                    acc[pp][ms][ns] = __builtin_amdgcn_wmma_f32_16x16x32_bf16(
                        false, af[ms], false, bfrg[pp][ns], (short)0,
                        acc[pp][ms][ns], false, false);

        if (kt + 1 < KT) {
#if HAVE_ASYNC_LDS
            async_wait0();
            __syncthreads();
#else
            *(uint4*)&As[buf ^ 1][ar][ac]      = pa0;
            *(uint4*)&As[buf ^ 1][ar + 64][ac] = pa1;
            *(uint4*)&Bs[buf ^ 1][0][ar][ac]   = pb0;
            if constexpr (NPROJ == 2) *(uint4*)&Bs[buf ^ 1][1][ar][ac] = pb1;
            __syncthreads();
#endif
        }
    }

    // ---- epilogue (C layout: VGPR i -> row i + 8*half-wave, col = lane&15)
    const int m_base = m0 + wm * 32;
    const int n_base = n0 + wn * 32;
    bf16*  OutB = (bf16*)Out;
    float* OutF = (float*)Out;
#pragma unroll
    for (int ms = 0; ms < 2; ++ms)
#pragma unroll
        for (int ns = 0; ns < 2; ++ns) {
            const int n = n_base + ns * 16 + lr;
            float bn = 0.f;
            if constexpr (MODE == 1 || MODE == 3) bn = bias[n];
#pragma unroll
            for (int i = 0; i < 8; ++i) {
                const int m = m_base + ms * 16 + hk * 8 + i;
                const size_t o = (size_t)m * Ncols + n;
                float a = acc[0][ms][ns][i];
                if constexpr (MODE == 0) {
                    float g = acc[1][ms][ns][i];
                    OutB[o] = (bf16)((a / (1.f + __expf(-a))) * g);
                } else if constexpr (MODE == 1) {
                    OutB[o] = (bf16)(1.f / (1.f + __expf(-(a + bn))));
                } else if constexpr (MODE == 2) {
                    float g = acc[1][ms][ns][i];
                    OutB[o] = (bf16)(a * (1.f / (1.f + __expf(-g))));
                } else {
                    OutF[o] = a + bn + resid[o];
                }
            }
        }
}

// ---------------------------------------------------------------- chunked scan
__global__ void scan_phase1(const bf16* __restrict__ f, const bf16* __restrict__ z,
                            float* __restrict__ P, float* __restrict__ Q) {
    const int idx = blockIdx.x * 256 + threadIdx.x;       // [0, NCHUNK*B*S)
    const int bs  = idx & (B_DIM * S_DIM - 1);
    const int c   = idx >> 13;                            // B*S = 8192 = 2^13
    const int b   = bs >> 11;                             // S = 2048 = 2^11
    const int s   = bs & (S_DIM - 1);
    const size_t base = ((size_t)b * T_DIM + (size_t)c * CLEN) * S_DIM + s;
    float p = 1.f, h = 0.f;
#pragma unroll 4
    for (int t = 0; t < CLEN; ++t) {
        const size_t off = base + (size_t)t * S_DIM;
        if (t + 16 < CLEN) {
            __builtin_prefetch(f + off + (size_t)16 * S_DIM, 0, 0);
            __builtin_prefetch(z + off + (size_t)16 * S_DIM, 0, 0);
        }
        float ft = (float)f[off];
        float zt = (float)z[off];
        p *= ft;
        h = (float)(bf16)(ft * h + zt);
    }
    P[idx] = p;
    Q[idx] = h;
}

__global__ void scan_phase2(const bf16* __restrict__ f, const bf16* __restrict__ z,
                            bf16* __restrict__ og,      // read og, write gated in place
                            const float* __restrict__ P, const float* __restrict__ Q) {
    const int idx = blockIdx.x * 256 + threadIdx.x;
    const int bs  = idx & (B_DIM * S_DIM - 1);
    const int c   = idx >> 13;
    const int b   = bs >> 11;
    const int s   = bs & (S_DIM - 1);
    float h = 0.f;
    for (int j = 0; j < c; ++j)
        h = P[j * (B_DIM * S_DIM) + bs] * h + Q[j * (B_DIM * S_DIM) + bs];
    const size_t base = ((size_t)b * T_DIM + (size_t)c * CLEN) * S_DIM + s;
#pragma unroll 4
    for (int t = 0; t < CLEN; ++t) {
        const size_t off = base + (size_t)t * S_DIM;
        if (t + 16 < CLEN) {
            __builtin_prefetch(f + off + (size_t)16 * S_DIM, 0, 0);
            __builtin_prefetch(z + off + (size_t)16 * S_DIM, 0, 0);
        }
        float ft = (float)f[off];
        float zt = (float)z[off];
        float gt = (float)og[off];
        h = (float)(bf16)(ft * h + zt);
        og[off] = (bf16)(gt * h);
    }
}

// ---------------------------------------------------------------- launch
extern "C" void kernel_launch(void* const* d_in, const int* in_sizes, int n_in,
                              void* d_out, int out_size, void* d_ws, size_t ws_size,
                              hipStream_t stream) {
    const float* x      = (const float*)d_in[0];
    const float* rms_w  = (const float*)d_in[1];
    const float* Wf_w   = (const float*)d_in[2];
    const float* Wf_b   = (const float*)d_in[3];
    const float* Wzg_w  = (const float*)d_in[4];
    const float* Wz_w   = (const float*)d_in[5];
    const float* Wga_w  = (const float*)d_in[6];
    const float* Wgb_w  = (const float*)d_in[7];
    const float* Wout_w = (const float*)d_in[8];
    const float* Wout_b = (const float*)d_in[9];
    float* out = (float*)d_out;

    const size_t MS = (size_t)M_DIM * S_DIM;       // 33.5M elems
    const size_t WE = (size_t)S_DIM * D_DIM;       // 4.19M elems per weight

    bf16* ws_xn = (bf16*)d_ws;
    bf16* ws_f  = ws_xn + MS;
    bf16* ws_z  = ws_f + MS;
    bf16* ws_og = ws_z + MS;                       // becomes `gated` in place
    bf16* w_f   = ws_og + MS;
    bf16* w_zg  = w_f  + WE;
    bf16* w_z   = w_zg + WE;
    bf16* w_ga  = w_z  + WE;
    bf16* w_gb  = w_ga + WE;
    bf16* w_out = w_gb + WE;
    float* ws_P = (float*)(w_out + WE);
    float* ws_Q = ws_P + (size_t)NCHUNK * B_DIM * S_DIM;

    const int cvtBlocks = (int)(WE / 1024);        // 4 elems/thread, 256 threads
    cvt_f32_bf16<<<cvtBlocks, 256, 0, stream>>>(Wf_w,   w_f);
    cvt_f32_bf16<<<cvtBlocks, 256, 0, stream>>>(Wzg_w,  w_zg);
    cvt_f32_bf16<<<cvtBlocks, 256, 0, stream>>>(Wz_w,   w_z);
    cvt_f32_bf16<<<cvtBlocks, 256, 0, stream>>>(Wga_w,  w_ga);
    cvt_f32_bf16<<<cvtBlocks, 256, 0, stream>>>(Wgb_w,  w_gb);
    cvt_f32_bf16<<<cvtBlocks, 256, 0, stream>>>(Wout_w, w_out);

    rmsnorm_kernel<<<M_DIM, 256, 0, stream>>>(x, rms_w, ws_xn);

    dim3 gemmGrid(S_DIM / 64, M_DIM / 128);        // (32, 128)
    // og = silu(xn Wga^T) * (xn Wgb^T)
    gemm_wmma<0, 2><<<gemmGrid, 256, 0, stream>>>(ws_xn, w_ga, w_gb, nullptr, nullptr,
                                                  ws_og, S_DIM, D_DIM);
    // f = sigmoid(xn Wf^T + b)
    gemm_wmma<1, 1><<<gemmGrid, 256, 0, stream>>>(ws_xn, w_f, nullptr, Wf_b, nullptr,
                                                  ws_f, S_DIM, D_DIM);
    // z = (xn Wz^T) * sigmoid(xn Wzg^T)
    gemm_wmma<2, 2><<<gemmGrid, 256, 0, stream>>>(ws_xn, w_z, w_zg, nullptr, nullptr,
                                                  ws_z, S_DIM, D_DIM);

    const int scanBlocks = (NCHUNK * B_DIM * S_DIM) / 256;   // 256
    scan_phase1<<<scanBlocks, 256, 0, stream>>>(ws_f, ws_z, ws_P, ws_Q);
    scan_phase2<<<scanBlocks, 256, 0, stream>>>(ws_f, ws_z, ws_og, ws_P, ws_Q);

    // y = gated Wout^T + b + x   (f32 out)
    gemm_wmma<3, 1><<<gemmGrid, 256, 0, stream>>>(ws_og, w_out, nullptr, Wout_b, x,
                                                  out, D_DIM, S_DIM);
}
